// MHSA_16131897164364
// MI455X (gfx1250) — matrix operand order repaired and checked by reading, not compiled, not run on hardware
//
#include <hip/hip_runtime.h>
#include <hip/hip_bf16.h>

#define DM 768
#define EM 2304
#define SEQ 1024
#define NB 8
#define NH 12
#define HD 64

#if __has_builtin(__builtin_amdgcn_global_load_async_to_lds_b128)
#define HAVE_ASYNC_LDS 1
#else
#define HAVE_ASYNC_LDS 0
#endif

typedef __attribute__((ext_vector_type(16))) _Float16 v16h;
typedef __attribute__((ext_vector_type(8)))  _Float16 half8;
typedef __attribute__((ext_vector_type(8)))  float    v8f;
typedef __attribute__((ext_vector_type(4)))  int      v4i;

// ---- WMMA operand loaders (this file's own contiguous k order, same for A and B) ----
// A-matrix 16x32 f16 (MxK): lane L<16 -> row L, halves[0..15]=K kb..kb+15,
// lanes>=16 same rows with kb = 16.
__device__ __forceinline__ v16h load_a16(const _Float16* __restrict__ base,
                                         int row0, int ld, int k0, int lane) {
  const int r  = lane & 15;
  const int kb = (lane >> 4) * 16;
  const _Float16* p = base + (size_t)(row0 + r) * ld + k0 + kb;
  half8 lo = *(const half8*)(p);
  half8 hi = *(const half8*)(p + 8);
  v16h out;
#pragma unroll
  for (int i = 0; i < 8; ++i) { out[i] = lo[i]; out[i + 8] = hi[i]; }
  return out;
}

// B-matrix 32x16 f16 (KxN), loaded as NxK: lane L -> col (L&15),
// halves[0..15] = 16 contiguous K values starting at (L>>4)*16.
__device__ __forceinline__ v16h load_b16(const _Float16* __restrict__ base,
                                         int col0, int ld, int k0, int lane) {
  const int n  = lane & 15;
  const int kb = (lane >> 4) * 16;
  const half8* p = (const half8*)(base + (size_t)(col0 + n) * ld + k0 + kb);
  half8 lo = p[0], hi = p[1];
  v16h out;
#pragma unroll
  for (int i = 0; i < 8; ++i) { out[i] = lo[i]; out[i + 8] = hi[i]; }
  return out;
}

__device__ __forceinline__ v8f wmma_f16(v16h a, v16h b, v8f c) {
  return __builtin_amdgcn_wmma_f32_16x16x32_f16(false, a, false, b, (short)0, c,
                                                false, false);
}

// ---- Kernel 1: x f32 -> f16 ----
__global__ void cvt_x_kernel(const float* __restrict__ x,
                             _Float16* __restrict__ xh, int n) {
  int i = blockIdx.x * blockDim.x + threadIdx.x;
  if (i < n) xh[i] = (_Float16)x[i];
}

// ---- Kernel 2: permute+convert W: wt[s*768 + h*64 + d][k] = W[k][(h*64+d)*3+s]
__global__ void cvt_w_kernel(const float* __restrict__ w,
                             _Float16* __restrict__ wt) {
  int i = blockIdx.x * blockDim.x + threadIdx.x;
  if (i >= DM * EM) return;
  int k = i / EM, e = i % EM;
  int hd = e / 3, s = e % 3;
  wt[((size_t)(s * DM + hd)) * DM + k] = (_Float16)w[i];
}

// ---- Kernel 3: QKV projection GEMM ----
// C[8192 x 2304] = xh @ wt^T (wt row-major = output-col-major operand).
// Q,K stored [B,H,N,64] token-major; V stored transposed [B,H,64,N].
__global__ __launch_bounds__(128) void qkv_gemm_kernel(
    const _Float16* __restrict__ xh, const _Float16* __restrict__ wt,
    const float* __restrict__ bias, _Float16* __restrict__ qh,
    _Float16* __restrict__ kh, _Float16* __restrict__ vt) {
  const int lane = threadIdx.x & 31;
  const int wave = threadIdx.x >> 5;
  const int m0 = blockIdx.x * 64 + wave * 16;  // token-row base for this wave
  const int n0 = blockIdx.y * 64;              // output-col base (one (s,h))
  const int s  = n0 / DM;
  const int h  = (n0 % DM) / HD;

  v8f acc[4] = {};
  for (int k0 = 0; k0 < DM; k0 += 32) {
    v16h a = load_a16(xh, m0, DM, k0, lane);
#pragma unroll
    for (int t = 0; t < 4; ++t)
      acc[t] = wmma_f16(a, load_b16(wt, n0 + t * 16, DM, k0, lane), acc[t]);
  }

  const int rhi = (lane >> 4) << 3;            // C layout: VGPR r -> M = r(+8)
  const int b = m0 >> 10;
  const int nbase = (m0 & (SEQ - 1)) + rhi;
#pragma unroll
  for (int t = 0; t < 4; ++t) {
    const int d = t * 16 + (lane & 15);
    const float bv = bias[(h * HD + d) * 3 + s];
    if (s == 2) {  // V: transposed store, tokens contiguous -> packed b128
      half8 pk;
#pragma unroll
      for (int r = 0; r < 8; ++r) pk[r] = (_Float16)(acc[t][r] + bv);
      *(half8*)(vt + ((size_t)((b * NH + h) * HD + d)) * SEQ + nbase) = pk;
    } else {       // Q/K: token-major store
      _Float16* dst = (s == 0) ? qh : kh;
#pragma unroll
      for (int r = 0; r < 8; ++r)
        dst[(((size_t)(b * NH + h)) * SEQ + (nbase + r)) * HD + d] =
            (_Float16)(acc[t][r] + bv);
    }
  }
}

// ---- Kernel 4: flash-style attention with async double-buffered K/V in LDS --
__global__ __launch_bounds__(128) void attn_kernel(
    const _Float16* __restrict__ qh, const _Float16* __restrict__ kh,
    const _Float16* __restrict__ vt, float* __restrict__ out) {
  __shared__ _Float16 Kb[2][64][64];    // key-major, 16 KB (double-buffered)
  __shared__ _Float16 Vb[2][64][64];    // d-major,   16 KB (double-buffered)
  __shared__ _Float16 ldsP[4][16][64];  // per-wave P staging, 8 KB

  const int lane = threadIdx.x & 31;
  const int wave = threadIdx.x >> 5;
  const int bh = blockIdx.x;
  const int b = bh / NH, h = bh % NH;
  const int q0 = blockIdx.y * 64 + wave * 16;
  const _Float16* Q = qh + (size_t)bh * SEQ * HD;
  const _Float16* K = kh + (size_t)bh * SEQ * HD;
  const _Float16* V = vt + (size_t)bh * HD * SEQ;
  const float scale = 0.03608439182435161f;  // 768^-0.5 (reference scales by D)

  // Stage one 64x64 f16 tile (64 rows x 128B) global -> LDS: 512 b128 chunks,
  // 4 per thread, via async-to-LDS (ASYNCcnt) when available.
  auto stage = [&](const _Float16* gbase, int gld, _Float16* lbase) {
#pragma unroll
    for (int i = 0; i < 4; ++i) {
      const int chunk = (int)threadIdx.x + i * 128;  // 0..511
      const int row = chunk >> 3, c8 = (chunk & 7) * 8;
      const _Float16* gs = gbase + (size_t)row * gld + c8;
      _Float16* ls = lbase + row * 64 + c8;
#if HAVE_ASYNC_LDS
      __builtin_amdgcn_global_load_async_to_lds_b128(
          (__attribute__((address_space(1))) v4i*)gs,
          (__attribute__((address_space(3))) v4i*)ls, 0, 0);
#else
      *(half8*)ls = *(const half8*)gs;
#endif
    }
  };
  auto wait_stage = [&]() {
#if HAVE_ASYNC_LDS
#if __has_builtin(__builtin_amdgcn_s_wait_asynccnt)
    __builtin_amdgcn_s_wait_asynccnt(0);
#else
    asm volatile("s_wait_asynccnt 0x0" ::: "memory");
#endif
#endif
    __syncthreads();
  };

  const v16h qa0 = load_a16(Q, q0, HD, 0, lane);
  const v16h qa1 = load_a16(Q, q0, HD, 32, lane);

  float mrow[8], lrow[8];
#pragma unroll
  for (int r = 0; r < 8; ++r) { mrow[r] = -1e30f; lrow[r] = 0.f; }
  v8f o[4] = {};

  int buf = 0;
  stage(K, HD, &Kb[0][0][0]);     // K block 0 is contiguous rows j..j+63
  stage(V, SEQ, &Vb[0][0][0]);    // V^T rows d, 128B each, stride SEQ
  wait_stage();

  for (int j = 0; j < SEQ; j += 64) {
    if (j + 64 < SEQ) {           // prefetch next key block into other buffer
      stage(K + (size_t)(j + 64) * HD, HD, &Kb[buf ^ 1][0][0]);
      stage(V + (j + 64), SEQ, &Vb[buf ^ 1][0][0]);
    }
    const int bn  = lane & 15;
    const int bkb = (lane >> 4) * 16;

    // S tile: 16 (queries) x 64 (keys), K operands from LDS
    v8f sc[4];
#pragma unroll
    for (int t = 0; t < 4; ++t) {
      v16h kb0, kb1;
#pragma unroll
      for (int i = 0; i < 16; ++i) {
        kb0[i] = Kb[buf][t * 16 + bn][bkb + i];
        kb1[i] = Kb[buf][t * 16 + bn][32 + bkb + i];
      }
      v8f c = {};
      c = wmma_f16(qa0, kb0, c);
      c = wmma_f16(qa1, kb1, c);
#pragma unroll
      for (int r = 0; r < 8; ++r) c[r] *= scale;
      sc[t] = c;
    }
    // online softmax stats (rows in VGPR components, cols across 16 lanes)
    float corr[8];
#pragma unroll
    for (int r = 0; r < 8; ++r) {
      float x = fmaxf(fmaxf(sc[0][r], sc[1][r]), fmaxf(sc[2][r], sc[3][r]));
      x = fmaxf(x, __shfl_xor(x, 1, 32));
      x = fmaxf(x, __shfl_xor(x, 2, 32));
      x = fmaxf(x, __shfl_xor(x, 4, 32));
      x = fmaxf(x, __shfl_xor(x, 8, 32));
      const float mnew = fmaxf(mrow[r], x);
      corr[r] = __expf(mrow[r] - mnew);
      mrow[r] = mnew;
    }
    const int rhi = (lane >> 4) << 3;
#pragma unroll
    for (int t = 0; t < 4; ++t) {
#pragma unroll
      for (int r = 0; r < 8; ++r) {
        const float p = __expf(sc[t][r] - mrow[r]);
        sc[t][r] = p;
        ldsP[wave][rhi + r][t * 16 + (lane & 15)] = (_Float16)p;  // C->LDS
      }
    }
#pragma unroll
    for (int r = 0; r < 8; ++r) {
      float ssum = sc[0][r] + sc[1][r] + sc[2][r] + sc[3][r];
      ssum += __shfl_xor(ssum, 1, 32);
      ssum += __shfl_xor(ssum, 2, 32);
      ssum += __shfl_xor(ssum, 4, 32);
      ssum += __shfl_xor(ssum, 8, 32);
      lrow[r] = lrow[r] * corr[r] + ssum;
    }
    asm volatile("s_wait_dscnt 0" ::: "memory");  // in-wave P visibility
    // re-read P in A-operand layout (packed 8-half LDS loads)
    const int rr = lane & 15;
    const int kb = (lane >> 4) * 16;
    v16h pa0, pa1;
#pragma unroll
    for (int i = 0; i < 8; ++i) {
      pa0[i]     = ldsP[wave][rr][kb + i];
      pa0[i + 8] = ldsP[wave][rr][8 + kb + i];
      pa1[i]     = ldsP[wave][rr][32 + kb + i];
      pa1[i + 8] = ldsP[wave][rr][40 + kb + i];
    }
    // O = O*corr + P @ V, V operands from LDS (d-major, contiguous keys)
#pragma unroll
    for (int dt = 0; dt < 4; ++dt) {
      v16h vb0, vb1;
#pragma unroll
      for (int i = 0; i < 16; ++i) {
        vb0[i] = Vb[buf][dt * 16 + bn][bkb + i];
        vb1[i] = Vb[buf][dt * 16 + bn][32 + bkb + i];
      }
#pragma unroll
      for (int r = 0; r < 8; ++r) o[dt][r] *= corr[r];
      o[dt] = wmma_f16(pa0, vb0, o[dt]);
      o[dt] = wmma_f16(pa1, vb1, o[dt]);
    }
    wait_stage();   // own async copies done + workgroup barrier -> swap
    buf ^= 1;
  }

  const int rhi = (lane >> 4) << 3;
#pragma unroll
  for (int dt = 0; dt < 4; ++dt) {
#pragma unroll
    for (int r = 0; r < 8; ++r) {
      const int m = q0 + rhi + r;
      out[((size_t)(b * SEQ + m)) * DM + h * HD + dt * 16 + (lane & 15)] =
          o[dt][r] / lrow[r];
    }
  }
}

extern "C" void kernel_launch(void* const* d_in, const int* in_sizes, int n_in,
                              void* d_out, int out_size, void* d_ws, size_t ws_size,
                              hipStream_t stream) {
  const float* x    = (const float*)d_in[0];
  const float* w    = (const float*)d_in[1];
  const float* bias = (const float*)d_in[2];
  float* out = (float*)d_out;

  char* ws = (char*)d_ws;                     // ~54 MB staged f16 workspace
  _Float16* xh = (_Float16*)(ws);             // 12,582,912 B
  _Float16* wt = (_Float16*)(ws + 12582912);  //  3,538,944 B
  _Float16* qh = (_Float16*)(ws + 16121856);  // 12,582,912 B
  _Float16* kh = (_Float16*)(ws + 28704768);  // 12,582,912 B
  _Float16* vt = (_Float16*)(ws + 41287680);  // 12,582,912 B (transposed V)

  const int nx = NB * SEQ * DM;
  cvt_x_kernel<<<(nx + 255) / 256, 256, 0, stream>>>(x, xh, nx);
  cvt_w_kernel<<<(DM * EM + 255) / 256, 256, 0, stream>>>(w, wt);
  qkv_gemm_kernel<<<dim3(8192 / 64, EM / 64), 128, 0, stream>>>(xh, wt, bias,
                                                                qh, kh, vt);
  attn_kernel<<<dim3(NB * NH, SEQ / 64), 128, 0, stream>>>(qh, kh, vt, out);
}
